// LossEC_WPG_88210038326255
// MI455X (gfx1250) — matrix operand order repaired and checked by reading, not compiled, not run on hardware
//
#include <hip/hip_runtime.h>
#include <hip/hip_bf16.h>
#include <math.h>

typedef __attribute__((ext_vector_type(16))) _Float16 v16h;
typedef __attribute__((ext_vector_type(8)))  _Float16 v8h;
typedef __attribute__((ext_vector_type(8)))  float    v8f;

#define DDIM 768
#define LDSP 776     // padded LDS row pitch in halves (1552B == 4 dwords mod 64 banks)
#define KT   20.0f   // temperature
#define THR  1.0f    // threshold

// ---------- helpers ----------
__device__ __forceinline__ void lse_combine(float& m, float& s, float m2, float s2) {
    float M = fmaxf(m, m2);
    s = s * __expf(m - M) + s2 * __expf(m2 - M);
    m = M;
}

__device__ __forceinline__ float block_reduce_sum256(float v, float* sh) {
    #pragma unroll
    for (int m = 16; m >= 1; m >>= 1) v += __shfl_xor(v, m, 32);
    int wid = threadIdx.x >> 5, lid = threadIdx.x & 31;
    __syncthreads();                 // protect sh reuse across calls
    if (lid == 0) sh[wid] = v;
    __syncthreads();
    float t = sh[0];
    #pragma unroll
    for (int w = 1; w < 8; w++) t += sh[w];
    return t;                        // every thread gets the total
}

__device__ __forceinline__ v16h load_fragA(const _Float16* base, int k0, int h) {
    // 16-bit A layout: lane half h: K = h*8..h*8+7 and 16+h*8..16+h*8+7
    v8h a0 = *(const v8h*)(base + k0 + h * 8);
    v8h a1 = *(const v8h*)(base + k0 + 16 + h * 8);
    return __builtin_shufflevector(a0, a1, 0,1,2,3,4,5,6,7,8,9,10,11,12,13,14,15);
}

__device__ __forceinline__ v16h load_fragB(const _Float16* base, int k0, int h) {
    // B 32x16 layout: lane half h: K = h*8..h*8+7 and 16+h*8..16+h*8+7 (same as A)
    v8h b0 = *(const v8h*)(base + k0 + h * 8);
    v8h b1 = *(const v8h*)(base + k0 + 16 + h * 8);
    return __builtin_shufflevector(b0, b1, 0,1,2,3,4,5,6,7,8,9,10,11,12,13,14,15);
}

// ---------- K1: L2-normalize rows -> f16 (objH for all 8*NB obj rows, argH for 3*NB text slots {0,2,3}) ----------
__global__ void k_normalize(const float* __restrict__ obj, const float* __restrict__ txt,
                            _Float16* __restrict__ objH, _Float16* __restrict__ argH, int NB) {
    int row = blockIdx.x;
    int nObjRows = NB * 8;
    const float* src;
    _Float16* dst;
    if (row < nObjRows) {
        src = obj + (size_t)row * DDIM;
        dst = objH + (size_t)row * DDIM;
    } else {
        int ar = row - nObjRows;
        int i = ar / 3, j = ar % 3;
        int s = (j == 0) ? 0 : (j + 1);   // slots 0,2,3
        src = txt + ((size_t)i * 4 + s) * DDIM;
        dst = argH + (size_t)ar * DDIM;
    }
    __shared__ float sh[8];
    float ss = 0.f;
    for (int d = threadIdx.x; d < DDIM; d += 256) { float v = src[d]; ss += v * v; }
    float tot = block_reduce_sum256(ss, sh);
    float sc = 1.0f / fmaxf(sqrtf(tot), 1e-12f);
    for (int d = threadIdx.x; d < DDIM; d += 256) dst[d] = (_Float16)(src[d] * sc);
}

// ---------- K2: diagonal-block max/argmax over 8 own objects (one wave per (i,j)) ----------
__global__ void k_diag(const _Float16* __restrict__ argH, const _Float16* __restrict__ objH,
                       float* __restrict__ diagmax, int* __restrict__ diagidx) {
    int gw = (blockIdx.x * blockDim.x + threadIdx.x) >> 5;   // (i*3+j)
    int lane = threadIdx.x & 31;
    int i = gw / 3;
    const _Float16* a = argH + (size_t)gw * DDIM;
    const _Float16* o = objH + (size_t)i * 8 * DDIM;
    float acc[8];
    #pragma unroll
    for (int k = 0; k < 8; k++) acc[k] = 0.f;
    for (int d = lane; d < DDIM; d += 32) {
        float av = (float)a[d];
        #pragma unroll
        for (int k = 0; k < 8; k++) acc[k] += av * (float)o[k * DDIM + d];
    }
    #pragma unroll
    for (int k = 0; k < 8; k++)
        #pragma unroll
        for (int m = 16; m >= 1; m >>= 1) acc[k] += __shfl_xor(acc[k], m, 32);
    if (lane == 0) {
        float best = acc[0]; int bi = 0;
        #pragma unroll
        for (int k = 1; k < 8; k++) if (acc[k] > best) { best = acc[k]; bi = k; }
        diagmax[gw] = best;   // un-scaled dot; max_value = KT*best
        diagidx[gw] = bi;
    }
}

// ---------- K3: build te/ve, renormalize, write f16 (one block per i) ----------
__global__ void k_teve(const float* __restrict__ obj, const float* __restrict__ txt,
                       const float* __restrict__ diagmax, const int* __restrict__ diagidx,
                       _Float16* __restrict__ teH, _Float16* __restrict__ veH) {
    int i = blockIdx.x;
    __shared__ float sh[8];
    __shared__ float scale[7];
    __shared__ float msk[3];
    __shared__ int   gidx[3];
    if (threadIdx.x < 3) {
        msk[threadIdx.x]  = (KT * diagmax[i * 3 + threadIdx.x] > THR) ? 1.f : 0.f;
        gidx[threadIdx.x] = diagidx[i * 3 + threadIdx.x];
    }
    __syncthreads();
    for (int s = 0; s < 7; s++) {
        const float* src = (s < 4) ? txt + ((size_t)i * 4 + s) * DDIM
                                   : obj + ((size_t)i * 8 + gidx[s - 4]) * DDIM;
        float ss = 0.f;
        for (int d = threadIdx.x; d < DDIM; d += 256) { float v = src[d]; ss += v * v; }
        float tot = block_reduce_sum256(ss, sh);
        if (threadIdx.x == 0) scale[s] = 1.0f / fmaxf(sqrtf(tot), 1e-12f);
        __syncthreads();
    }
    const float* t0 = txt + (size_t)i * 4 * DDIM;
    const float* o0 = obj + (size_t)i * 8 * DDIM;
    float tev[3], vev[3], teSS = 0.f, veSS = 0.f;
    #pragma unroll
    for (int t = 0; t < 3; t++) {
        int d = threadIdx.x + t * 256;
        float te = t0[1 * DDIM + d] * scale[1];
        te += msk[0] * t0[0 * DDIM + d] * scale[0];
        te += msk[1] * t0[2 * DDIM + d] * scale[2];
        te += msk[2] * t0[3 * DDIM + d] * scale[3];
        float ve = msk[0] * o0[gidx[0] * DDIM + d] * scale[4]
                 + msk[1] * o0[gidx[1] * DDIM + d] * scale[5]
                 + msk[2] * o0[gidx[2] * DDIM + d] * scale[6];
        tev[t] = te; vev[t] = ve;
        teSS += te * te; veSS += ve * ve;
    }
    float teT = block_reduce_sum256(teSS, sh);
    float veT = block_reduce_sum256(veSS, sh);
    float tsc = 1.0f / fmaxf(sqrtf(teT), 1e-12f);
    float vsc = 1.0f / fmaxf(sqrtf(veT), 1e-12f);
    #pragma unroll
    for (int t = 0; t < 3; t++) {
        int d = threadIdx.x + t * 256;
        teH[(size_t)i * DDIM + d] = (_Float16)(tev[t] * tsc);
        veH[(size_t)i * DDIM + d] = (_Float16)(vev[t] * vsc);
    }
}

// ---------- K4: EC gemm Z = KT * te @ ve^T  (one 16x16 WMMA tile per wave) ----------
__global__ void k_ecgemm(const _Float16* __restrict__ teH, const _Float16* __restrict__ veH,
                         float* __restrict__ Z, int NB) {
    int gw = (blockIdx.x * blockDim.x + threadIdx.x) >> 5;
    int lane = threadIdx.x & 31;
    int ntile = NB >> 4;
    int tm = gw / ntile, tn = gw % ntile;
    int h = lane >> 4, ln = lane & 15;
    const _Float16* ap = teH + (size_t)(tm * 16 + ln) * DDIM;
    const _Float16* bp = veH + (size_t)(tn * 16 + ln) * DDIM;
    v8f c = {};
    for (int kc = 0; kc < DDIM / 32; kc++) {
        int k0 = kc * 32;
        v16h A = load_fragA(ap, k0, h);
        v16h Bv = load_fragB(bp, k0, h);
        c = __builtin_amdgcn_wmma_f32_16x16x32_f16(false, A, false, Bv, (short)0, c, false, false);
    }
    #pragma unroll
    for (int r = 0; r < 8; r++)
        Z[(size_t)(tm * 16 + h * 8 + r) * NB + tn * 16 + ln] = KT * c[r];
}

// ---------- K6: big GEMM with fused streaming log-sum-exp.
// Block = 8 waves owning 128 arg rows; B panel (64 cols x 768 k, f16) cooperatively
// staged in LDS once per column chunk and shared by all 8 waves (8x less L2 traffic).
// wpgterm[r] = LSE_r - KT*diagmax[r]
__global__ void __launch_bounds__(256) k_wpg(const _Float16* __restrict__ argH,
                                             const _Float16* __restrict__ objH,
                                             const float* __restrict__ diagmax,
                                             float* __restrict__ wpgterm, int NB) {
    extern __shared__ _Float16 ldsB[];                       // 64 * LDSP halves (~97 KB)
    int w    = threadIdx.x >> 5;
    int lane = threadIdx.x & 31;
    int h = lane >> 4, ln = lane & 15;
    int rowBase = blockIdx.x * 128 + w * 16;
    int nCols = NB * 8;
    const _Float16* ap = argH + (size_t)(rowBase + ln) * DDIM;

    float mx[8], sm[8];
    #pragma unroll
    for (int r = 0; r < 8; r++) { mx[r] = -1e30f; sm[r] = 0.f; }

    for (int nb = 0; nb < nCols; nb += 64) {
        __syncthreads();   // previous chunk fully consumed before overwrite
        // cooperative stage of B panel: 64 cols x 768 halves = 6144 x v8h, 24 per thread
        #pragma unroll
        for (int it = 0; it < 24; it++) {
            int f   = threadIdx.x + it * 256;   // 0..6143
            int col = f / 96;                    // 96 v8h segments per column
            int seg = f % 96;
            *(v8h*)(ldsB + col * LDSP + seg * 8) =
                *(const v8h*)(objH + (size_t)(nb + col) * DDIM + seg * 8);
        }
        __syncthreads();

        v8f c[4];
        #pragma unroll
        for (int t = 0; t < 4; t++) c[t] = (v8f){};
        for (int kc = 0; kc < DDIM / 32; kc++) {
            int k0 = kc * 32;
            v16h A = load_fragA(ap, k0, h);
            #pragma unroll
            for (int t = 0; t < 4; t++) {
                v16h Bv = load_fragB(ldsB + (t * 16 + ln) * LDSP, k0, h);
                c[t] = __builtin_amdgcn_wmma_f32_16x16x32_f16(false, A, false, Bv,
                                                              (short)0, c[t], false, false);
            }
        }
        // streaming LSE update: lane holds column (nb + t*16 + ln), rows h*8 + r
        #pragma unroll
        for (int t = 0; t < 4; t++)
            #pragma unroll
            for (int r = 0; r < 8; r++) {
                float v = KT * c[t][r];
                if (v > mx[r]) { sm[r] = sm[r] * __expf(mx[r] - v) + 1.0f; mx[r] = v; }
                else           { sm[r] += __expf(v - mx[r]); }
            }
    }
    // combine across the 16 lanes sharing the same row set (xor masks < 16 stay in half)
    #pragma unroll
    for (int m = 8; m >= 1; m >>= 1)
        #pragma unroll
        for (int r = 0; r < 8; r++) {
            float om = __shfl_xor(mx[r], m, 32);
            float os = __shfl_xor(sm[r], m, 32);
            lse_combine(mx[r], sm[r], om, os);
        }
    if (ln == 0) {
        #pragma unroll
        for (int r = 0; r < 8; r++) {
            int row = rowBase + h * 8 + r;
            wpgterm[row] = (mx[r] + __logf(sm[r])) - KT * diagmax[row];
        }
    }
}

// ---------- K5: generic per-row LSE over an f32 matrix (wave per row) ----------
__global__ void k_lse(const float* __restrict__ Z, float* __restrict__ out,
                      int ncols, int rowpitch, int colstride) {
    int row = (blockIdx.x * blockDim.x + threadIdx.x) >> 5;
    int lane = threadIdx.x & 31;
    float m = -1e30f, s = 0.f;
    for (int j = lane; j < ncols; j += 32) {
        float v = Z[(size_t)row * rowpitch + (size_t)j * colstride];
        if (v > m) { s = s * __expf(m - v) + 1.0f; m = v; }
        else       { s += __expf(v - m); }
    }
    #pragma unroll
    for (int k = 16; k >= 1; k >>= 1) {
        float om = __shfl_xor(m, k, 32);
        float os = __shfl_xor(s, k, 32);
        lse_combine(m, s, om, os);
    }
    if (lane == 0) out[row] = m + __logf(s);
}

// ---------- EC per-row term ----------
__global__ void k_ecterm(const float* __restrict__ Z, const float* __restrict__ lseR,
                         const float* __restrict__ lseC, float* __restrict__ ecterm, int NB) {
    int i = blockIdx.x * blockDim.x + threadIdx.x;
    if (i < NB) ecterm[i] = 0.5f * (lseR[i] + lseC[i]) - Z[(size_t)i * NB + i];
}

// ---------- verb CE per-row term (wave per row) ----------
__global__ void k_verb(const float* __restrict__ preds, const int* __restrict__ labels,
                       float* __restrict__ vterm, int C) {
    int row = (blockIdx.x * blockDim.x + threadIdx.x) >> 5;
    int lane = threadIdx.x & 31;
    const float* p = preds + (size_t)row * C;
    float m = -1e30f, s = 0.f;
    for (int j = lane; j < C; j += 32) {
        float v = p[j];
        if (v > m) { s = s * __expf(m - v) + 1.0f; m = v; }
        else       { s += __expf(v - m); }
    }
    #pragma unroll
    for (int k = 16; k >= 1; k >>= 1) {
        float om = __shfl_xor(m, k, 32);
        float os = __shfl_xor(s, k, 32);
        lse_combine(m, s, om, os);
    }
    if (lane == 0) vterm[row] = (m + __logf(s)) - p[labels[row]];
}

// ---------- final weighted mean ----------
__global__ void k_final(const float* __restrict__ wpgterm, const float* __restrict__ vterm,
                        const float* __restrict__ ecterm, float* __restrict__ out, int NB) {
    __shared__ float sh[8];
    float s1 = 0.f, s2 = 0.f, s3 = 0.f;
    for (int r = threadIdx.x; r < 3 * NB; r += 256) s1 += wpgterm[r];
    for (int r = threadIdx.x; r < NB;     r += 256) s2 += vterm[r];
    for (int r = threadIdx.x; r < NB;     r += 256) s3 += ecterm[r];
    s1 = block_reduce_sum256(s1, sh);
    s2 = block_reduce_sum256(s2, sh);
    s3 = block_reduce_sum256(s3, sh);
    if (threadIdx.x == 0)
        out[0] = 0.3f * (s1 / (3.0f * NB)) + 0.2f * (s2 / NB) + 0.5f * (s3 / NB);
}

// ---------- launcher ----------
extern "C" void kernel_launch(void* const* d_in, const int* in_sizes, int n_in,
                              void* d_out, int out_size, void* d_ws, size_t ws_size,
                              hipStream_t stream) {
    const float* obj   = (const float*)d_in[0];   // [B,5,8,768]
    const float* txt   = (const float*)d_in[1];   // [B,5,4,768]
    const float* preds = (const float*)d_in[2];   // [B,5,C]
    const int*   lbls  = (const int*)d_in[3];     // [B,5]
    float* out = (float*)d_out;

    const int B  = in_sizes[0] / (5 * 8 * DDIM);
    const int NB = 5 * B;                          // 1280
    const int C  = in_sizes[2] / NB;               // 504

    char* ws = (char*)d_ws;
    size_t off = 0;
    auto alloc = [&](size_t bytes) { void* p = ws + off; off = (off + bytes + 255) & ~(size_t)255; return p; };
    _Float16* objH   = (_Float16*)alloc((size_t)NB * 8 * DDIM * 2);
    _Float16* argH   = (_Float16*)alloc((size_t)NB * 3 * DDIM * 2);
    _Float16* teH    = (_Float16*)alloc((size_t)NB * DDIM * 2);
    _Float16* veH    = (_Float16*)alloc((size_t)NB * DDIM * 2);
    float*    Z      = (float*)   alloc((size_t)NB * NB * 4);
    float*    dmax   = (float*)   alloc((size_t)NB * 3 * 4);
    int*      didx   = (int*)     alloc((size_t)NB * 3 * 4);
    float*    lseR   = (float*)   alloc((size_t)NB * 4);
    float*    lseC   = (float*)   alloc((size_t)NB * 4);
    float*    wpgt   = (float*)   alloc((size_t)NB * 3 * 4);
    float*    vt     = (float*)   alloc((size_t)NB * 4);
    float*    ect    = (float*)   alloc((size_t)NB * 4);

    // 1) normalize -> f16 copies
    k_normalize<<<NB * 8 + NB * 3, 256, 0, stream>>>(obj, txt, objH, argH, NB);
    // 2) diagonal block max/argmax
    k_diag<<<(NB * 3) / 8, 256, 0, stream>>>(argH, objH, dmax, didx);
    // 3) te/ve
    k_teve<<<NB, 256, 0, stream>>>(obj, txt, dmax, didx, teH, veH);
    // 4) EC gemm (WMMA)
    k_ecgemm<<<((NB / 16) * (NB / 16)) / 8, 256, 0, stream>>>(teH, veH, Z, NB);
    // 5) EC row/col LSE
    k_lse<<<NB / 8, 256, 0, stream>>>(Z, lseR, NB, NB, 1);
    k_lse<<<NB / 8, 256, 0, stream>>>(Z, lseC, NB, 1, NB);
    k_ecterm<<<(NB + 255) / 256, 256, 0, stream>>>(Z, lseR, lseC, ect, NB);
    // 6) big GEMM + fused LSE (WMMA, LDS-staged B panel)
    const size_t ldsBytes = (size_t)64 * LDSP * sizeof(_Float16);   // ~97 KB
    k_wpg<<<(NB * 3) / 128, 256, ldsBytes, stream>>>(argH, objH, dmax, wpgt, NB);
    // 7) verb CE
    k_verb<<<NB / 8, 256, 0, stream>>>(preds, lbls, vt, C);
    // 8) combine
    k_final<<<1, 256, 0, stream>>>(wpgt, vt, ect, out, NB);
}
